// LengthRegulator_42365557407777
// MI455X (gfx1250) — matrix operand order repaired and checked by reading, not compiled, not run on hardware
//
#include <hip/hip_runtime.h>

#define B_ 64
#define L_ 256
#define C_ 384
#define K_ 1152   // 3*C

typedef __attribute__((ext_vector_type(16))) _Float16 v16h;
typedef __attribute__((ext_vector_type(8)))  _Float16 v8h;
typedef __attribute__((ext_vector_type(8)))  float    v8f;

// ---------------- prep kernels ----------------

__global__ void cvt_f32_f16(const float* __restrict__ in, _Float16* __restrict__ out, int n) {
    int i = blockIdx.x * blockDim.x + threadIdx.x;
    if (i < n) out[i] = (_Float16)in[i];
}

// conv w: [Cout, Cin, 3] f32  ->  wt: [Cout, K] f16 with k = tap*C + cin
__global__ void prep_w(const float* __restrict__ w, _Float16* __restrict__ wt) {
    int i = blockIdx.x * blockDim.x + threadIdx.x;
    if (i < C_ * C_ * 3) {
        int tap  = i % 3;
        int cin  = (i / 3) % C_;
        int cout = i / (3 * C_);
        wt[(size_t)cout * K_ + tap * C_ + cin] = (_Float16)w[i];
    }
}

// inclusive scan of durations per batch row
__global__ void scan_dur(const int* __restrict__ tgt, int* __restrict__ cum) {
    __shared__ int s[L_];
    int b = blockIdx.x, t = threadIdx.x;
    s[t] = tgt[b * L_ + t];
    __syncthreads();
    for (int off = 1; off < L_; off <<= 1) {
        int v = (t >= off) ? s[t - off] : 0;
        __syncthreads();
        s[t] += v;
        __syncthreads();
    }
    cum[b * L_ + t] = s[t];
}

// ---------------- fused conv3 (WMMA GEMM) + bias + ReLU + LayerNorm ----------------
// Workgroup: 256 threads (8 waves). One 16-row M-tile x full N=384.
// Each wave owns 3 N-tiles. K loop: 36 steps of 32 (each step inside one tap).

template <bool FINAL>
__global__ __launch_bounds__(256) void conv_ln(
    const _Float16* __restrict__ inh,     // [B*L, C] f16
    const _Float16* __restrict__ wt,      // [C, K]  f16 (N-major, transposed)
    const float* __restrict__ bias,       // [C]
    const float* __restrict__ g,          // ln gamma [C]
    const float* __restrict__ be,         // ln beta  [C]
    _Float16* __restrict__ outh,          // layer1 out: [B*L, C] f16
    const float* __restrict__ lin_w,      // [C]   (FINAL only)
    const float* __restrict__ lin_b,      // [1]   (FINAL only)
    float* __restrict__ log_dur)          // [B*L] (FINAL only)
{
    __shared__ float tile[16][C_];        // 24 KB post-ReLU activations
    __shared__ float red[16][16][2];
    __shared__ float stats[16][2];        // mean, inv_std

    const int tid  = threadIdx.x;
    const int lane = tid & 31;
    const int wv   = tid >> 5;            // 0..7
    const int nloc = lane & 15;
    const int half = lane >> 4;           // 0/1

    const int mtile = blockIdx.x;         // 0..1023 (B*L/16)
    const int b     = (mtile * 16) / L_;
    const int lbase = (mtile * 16) % L_;  // tiles never straddle a batch (256 % 16 == 0)

    v8f acc0 = {}, acc1 = {}, acc2 = {};
    const int n0 = (wv * 3 + 0) * 16 + nloc;
    const int n1 = (wv * 3 + 1) * 16 + nloc;
    const int n2 = (wv * 3 + 2) * 16 + nloc;

    for (int kk = 0; kk < 36; ++kk) {
        const int tap  = kk / 12;
        const int cinb = (kk - tap * 12) * 32;
        const int kofs = kk * 32 + half * 16;

        // ---- A fragment (16x32 f16): lane = row (nloc); two 16B loads ----
        v16h a;
        const int lp = lbase + nloc - 1 + tap;   // conv 'same' pad
        if (lp >= 0 && lp < L_) {
            const _Float16* ap = inh + ((size_t)(b * L_ + lp) * C_ + cinb + half * 16);
            v8h lo = *(const v8h*)ap;
            v8h hi = *(const v8h*)(ap + 8);
            #pragma unroll
            for (int i = 0; i < 8; ++i) { a[i] = lo[i]; a[8 + i] = hi[i]; }
        } else {
            #pragma unroll
            for (int i = 0; i < 16; ++i) a[i] = (_Float16)0.0f;
        }

        // ---- B fragments (32x16 f16): lane = col; 16 contiguous f16 along K ----
        v16h bf;
        {
            const _Float16* bp = wt + (size_t)n0 * K_ + kofs;
            v8h lo = *(const v8h*)bp; v8h hi = *(const v8h*)(bp + 8);
            #pragma unroll
            for (int i = 0; i < 8; ++i) { bf[i] = lo[i]; bf[8 + i] = hi[i]; }
        }
        acc0 = __builtin_amdgcn_wmma_f32_16x16x32_f16(false, a, false, bf, (short)0, acc0, false, false);
        {
            const _Float16* bp = wt + (size_t)n1 * K_ + kofs;
            v8h lo = *(const v8h*)bp; v8h hi = *(const v8h*)(bp + 8);
            #pragma unroll
            for (int i = 0; i < 8; ++i) { bf[i] = lo[i]; bf[8 + i] = hi[i]; }
        }
        acc1 = __builtin_amdgcn_wmma_f32_16x16x32_f16(false, a, false, bf, (short)0, acc1, false, false);
        {
            const _Float16* bp = wt + (size_t)n2 * K_ + kofs;
            v8h lo = *(const v8h*)bp; v8h hi = *(const v8h*)(bp + 8);
            #pragma unroll
            for (int i = 0; i < 8; ++i) { bf[i] = lo[i]; bf[8 + i] = hi[i]; }
        }
        acc2 = __builtin_amdgcn_wmma_f32_16x16x32_f16(false, a, false, bf, (short)0, acc2, false, false);
    }

    // ---- bias + ReLU -> LDS tile (C/D layout: VGPR r = row half*8+r, lane%16 = col) ----
    {
        const float bv0 = bias[n0], bv1 = bias[n1], bv2 = bias[n2];
        #pragma unroll
        for (int r = 0; r < 8; ++r) {
            float v0 = acc0[r] + bv0; tile[half * 8 + r][n0] = v0 > 0.f ? v0 : 0.f;
            float v1 = acc1[r] + bv1; tile[half * 8 + r][n1] = v1 > 0.f ? v1 : 0.f;
            float v2 = acc2[r] + bv2; tile[half * 8 + r][n2] = v2 > 0.f ? v2 : 0.f;
        }
    }
    __syncthreads();

    // ---- LayerNorm stats: 16 rows x 16 partial-threads, 24 elems each ----
    const int row = tid >> 4;
    const int sub = tid & 15;
    {
        float s1 = 0.f, s2 = 0.f;
        const int c0 = sub * 24;
        #pragma unroll 4
        for (int c = c0; c < c0 + 24; ++c) { float v = tile[row][c]; s1 += v; s2 += v * v; }
        red[row][sub][0] = s1; red[row][sub][1] = s2;
    }
    __syncthreads();
    if (sub == 0) {
        float t1 = 0.f, t2 = 0.f;
        #pragma unroll
        for (int i = 0; i < 16; ++i) { t1 += red[row][i][0]; t2 += red[row][i][1]; }
        float mean = t1 * (1.0f / C_);
        float var  = t2 * (1.0f / C_) - mean * mean;
        stats[row][0] = mean;
        stats[row][1] = rsqrtf(var + 1e-5f);
    }
    __syncthreads();

    if constexpr (!FINAL) {
        // normalize + write f16 hidden for conv2
        for (int i = tid; i < 16 * C_; i += 256) {
            int r = i / C_, c = i - r * C_;
            float v = (tile[r][c] - stats[r][0]) * stats[r][1] * g[c] + be[c];
            outh[(size_t)(b * L_ + lbase + r) * C_ + c] = (_Float16)v;
        }
    } else {
        // fused final linear (C->1) + ReLU -> log_dur
        float d = 0.f;
        const int c0 = sub * 24;
        #pragma unroll 4
        for (int c = c0; c < c0 + 24; ++c) {
            float v = (tile[row][c] - stats[row][0]) * stats[row][1] * g[c] + be[c];
            d += v * lin_w[c];
        }
        red[row][sub][0] = d;
        __syncthreads();
        if (sub == 0) {
            float t = lin_b[0];
            #pragma unroll
            for (int i = 0; i < 16; ++i) t += red[row][i][0];
            log_dur[b * L_ + lbase + row] = t > 0.f ? t : 0.f;
        }
    }
}

// ---------------- length regulator: per-row binary search + 384-float row copy ----------------

__global__ __launch_bounds__(256) void length_reg(
    const float* __restrict__ x, const int* __restrict__ cum,
    float* __restrict__ out, int T, int nrows)
{
    const int lane  = threadIdx.x & 31;
    const int wv    = threadIdx.x >> 5;
    const int rowid = blockIdx.x * 8 + wv;
    if (rowid >= nrows) return;
    const int b = rowid / T;
    const int t = rowid - b * T;
    const int* cb = cum + b * L_;
    float4* dst = (float4*)(out + (size_t)rowid * C_);
    if (t < cb[L_ - 1]) {
        int lo = 0, hi = L_ - 1;                   // first j with cum[j] > t
        while (lo < hi) { int mid = (lo + hi) >> 1; if (cb[mid] > t) hi = mid; else lo = mid + 1; }
        const float4* src = (const float4*)(x + ((size_t)b * L_ + lo) * C_);
        for (int i = lane; i < C_ / 4; i += 32) dst[i] = src[i];
    } else {
        float4 z = make_float4(0.f, 0.f, 0.f, 0.f);
        for (int i = lane; i < C_ / 4; i += 32) dst[i] = z;
    }
}

// ---------------- launch ----------------

extern "C" void kernel_launch(void* const* d_in, const int* in_sizes, int n_in,
                              void* d_out, int out_size, void* d_ws, size_t ws_size,
                              hipStream_t stream) {
    const float* x      = (const float*)d_in[0];
    const int*   target = (const int*)d_in[1];
    // d_in[2] = mel_max_length (device scalar); derive T from out_size instead.
    const float* c1w = (const float*)d_in[3];
    const float* c1b = (const float*)d_in[4];
    const float* g1  = (const float*)d_in[5];
    const float* b1  = (const float*)d_in[6];
    const float* c2w = (const float*)d_in[7];
    const float* c2b = (const float*)d_in[8];
    const float* g2  = (const float*)d_in[9];
    const float* b2  = (const float*)d_in[10];
    const float* lw  = (const float*)d_in[11];
    const float* lb  = (const float*)d_in[12];

    const int T = (out_size - B_ * L_) / (B_ * C_);   // 2560

    // workspace layout (all 256B-aligned)
    char* p = (char*)d_ws;
    _Float16* xh  = (_Float16*)p; p += (size_t)B_ * L_ * C_ * 2;   // 12.6 MB
    _Float16* h1h = (_Float16*)p; p += (size_t)B_ * L_ * C_ * 2;   // 12.6 MB
    _Float16* w1t = (_Float16*)p; p += (size_t)C_ * K_ * 2;        // 0.88 MB
    _Float16* w2t = (_Float16*)p; p += (size_t)C_ * K_ * 2;        // 0.88 MB
    int*      cum = (int*)p;

    float* out     = (float*)d_out;
    float* log_dur = out + (size_t)B_ * T * C_;

    const int nElem = B_ * L_ * C_;
    cvt_f32_f16<<<(nElem + 255) / 256, 256, 0, stream>>>(x, xh, nElem);
    prep_w<<<(C_ * C_ * 3 + 255) / 256, 256, 0, stream>>>(c1w, w1t);
    prep_w<<<(C_ * C_ * 3 + 255) / 256, 256, 0, stream>>>(c2w, w2t);
    scan_dur<<<B_, L_, 0, stream>>>(target, cum);

    const int mtiles = B_ * L_ / 16;   // 1024
    conv_ln<false><<<mtiles, 256, 0, stream>>>(xh,  w1t, c1b, g1, b1, h1h, nullptr, nullptr, nullptr);
    conv_ln<true ><<<mtiles, 256, 0, stream>>>(h1h, w2t, c2b, g2, b2, nullptr, lw, lb, log_dur);

    const int nrows = B_ * T;
    length_reg<<<(nrows + 7) / 8, 256, 0, stream>>>(x, cum, out, T, nrows);
}
